// build_transformer_local_60000693125642
// MI455X (gfx1250) — matrix operand order repaired and checked by reading, not compiled, not run on hardware
//
#include <hip/hip_runtime.h>

typedef __attribute__((ext_vector_type(16))) __bf16 v16bf;
typedef __attribute__((ext_vector_type(8)))  float  v8f;

#define EPSN 1e-12f
constexpr int Bq = 16, Np = 128, Cc = 768, Gg = 2000, CLSK = 100, TOPK = 10;

// ---------------- query CLS inverse norms ----------------
__global__ void qcinv_kernel(const float* __restrict__ qc, float* __restrict__ qcinv) {
  __shared__ float red[256];
  int b = blockIdx.x, t = threadIdx.x;
  float s = 0.f;
  for (int c = t; c < Cc; c += 256) { float v = qc[b * Cc + c]; s += v * v; }
  red[t] = s; __syncthreads();
  for (int st = 128; st > 0; st >>= 1) { if (t < st) red[t] += red[t + st]; __syncthreads(); }
  if (t == 0) qcinv[b] = 1.f / fmaxf(sqrtf(red[0]), EPSN);
}

// ---------------- CLS distance matrix [B][G] ----------------
__global__ void cls_dist_kernel(const float* __restrict__ gcls, const float* __restrict__ qc,
                                const float* __restrict__ qcinv, float* __restrict__ dall) {
  __shared__ float red[256];
  __shared__ float grow[Cc];
  __shared__ float ginv_s;
  int g = blockIdx.x, t = threadIdx.x;
  const float* gr = gcls + (size_t)g * Cc;
  float s = 0.f;
  for (int c = t; c < Cc; c += 256) { float v = gr[c]; grow[c] = v; s += v * v; }
  red[t] = s; __syncthreads();
  for (int st = 128; st > 0; st >>= 1) { if (t < st) red[t] += red[t + st]; __syncthreads(); }
  if (t == 0) ginv_s = 1.f / fmaxf(sqrtf(red[0]), EPSN);
  __syncthreads();
  float gi = ginv_s;
  for (int b = 0; b < Bq; b++) {
    float d = 0.f;
    for (int c = t; c < Cc; c += 256) d += grow[c] * qc[b * Cc + c];
    red[t] = d; __syncthreads();
    for (int st = 128; st > 0; st >>= 1) { if (t < st) red[t] += red[t + st]; __syncthreads(); }
    if (t == 0) dall[b * Gg + g] = red[0] * gi * qcinv[b];
    __syncthreads();
  }
}

// ---------------- normalize query patches -> bf16 ----------------
__global__ void qnorm_kernel(const float* __restrict__ qp, __bf16* __restrict__ Qn) {
  __shared__ float red[256];
  __shared__ float invs;
  int r = blockIdx.x;  // b*Np + n
  int t = threadIdx.x;
  const float* src = qp + (size_t)r * Cc;
  float s = 0.f;
  for (int c = t; c < Cc; c += 256) { float v = src[c]; s += v * v; }
  red[t] = s; __syncthreads();
  for (int st = 128; st > 0; st >>= 1) { if (t < st) red[t] += red[t + st]; __syncthreads(); }
  if (t == 0) invs = 1.f / fmaxf(sqrtf(red[0]), EPSN);
  __syncthreads();
  float iv = invs;
  for (int c = t; c < Cc; c += 256) Qn[(size_t)r * Cc + c] = (__bf16)(src[c] * iv);
}

// ---------------- top-100 CLS candidates per sample ----------------
__global__ void topk_cls_kernel(const float* __restrict__ dall, int* __restrict__ clsIdx) {
  __shared__ float sc[Gg];
  __shared__ float rv[256];
  __shared__ int   ri[256];
  int b = blockIdx.x, t = threadIdx.x;
  for (int g = t; g < Gg; g += 256) sc[g] = dall[b * Gg + g];
  __syncthreads();
  for (int k = 0; k < CLSK; k++) {
    float bv = -3.4e38f; int bi = 0x7fffffff;
    for (int g = t; g < Gg; g += 256) {
      float v = sc[g];
      if (v > bv || (v == bv && g < bi)) { bv = v; bi = g; }
    }
    rv[t] = bv; ri[t] = bi; __syncthreads();
    for (int st = 128; st > 0; st >>= 1) {
      if (t < st) {
        float ov = rv[t + st]; int oi = ri[t + st];
        if (ov > rv[t] || (ov == rv[t] && oi < ri[t])) { rv[t] = ov; ri[t] = oi; }
      }
      __syncthreads();
    }
    if (t == 0) { clsIdx[b * CLSK + k] = ri[0]; sc[ri[0]] = -3.4e38f; }
    __syncthreads();
  }
}

// ---------------- EMD score per (b, candidate) via WMMA ----------------
__global__ __launch_bounds__(256) void emd_kernel(
    const float* __restrict__ gpatch, const __bf16* __restrict__ Qn,
    const float* __restrict__ dall, const int* __restrict__ clsIdx,
    float* __restrict__ dist) {
  constexpr int SP = 40;  // padded LDS row stride (halfs) -> conflict-free b128
  __shared__ __bf16 Qs[128 * SP];
  __shared__ __bf16 Gs[128 * SP];
  __shared__ float rowsq[128];
  __shared__ float colpart[8][128];
  __shared__ float cm[128];
  __shared__ float sumRow;

  int kk = blockIdx.x, b = blockIdx.y;
  int t = threadIdx.x;
  int g = clsIdx[b * CLSK + kk];
  int lane = t & 31, w = t >> 5;

  if (t < 128) rowsq[t] = 0.f;
  if (t == 0) sumRow = 0.f;

  const __bf16* Qbase = Qn + (size_t)b * Np * Cc;
  const float*  Gbase = gpatch + (size_t)g * Np * Cc;

  int srow = t >> 1;         // staging row 0..127
  int scol = (t & 1) * 16;   // 0 or 16 within the 32-wide chunk

  v8f acc[8] = {};
  __syncthreads();

  for (int kb = 0; kb < Cc; kb += 32) {
    // stage Q chunk (already bf16, normalized)
    {
      const uint4* src = (const uint4*)(Qbase + (size_t)srow * Cc + kb + scol);
      uint4 a0 = src[0], a1 = src[1];
      uint4* dst = (uint4*)(&Qs[srow * SP + scol]);
      dst[0] = a0; dst[1] = a1;
    }
    // stage G chunk fp32 -> bf16 + accumulate row sum-of-squares
    {
      const float4* src = (const float4*)(Gbase + (size_t)srow * Cc + kb + scol);
      float4 f0 = src[0], f1 = src[1], f2 = src[2], f3 = src[3];
      float fv[16] = {f0.x, f0.y, f0.z, f0.w, f1.x, f1.y, f1.z, f1.w,
                      f2.x, f2.y, f2.z, f2.w, f3.x, f3.y, f3.z, f3.w};
      float ss = 0.f;
      union { __bf16 h[16]; uint4 q[2]; } u;
#pragma unroll
      for (int i = 0; i < 16; i++) { ss += fv[i] * fv[i]; u.h[i] = (__bf16)fv[i]; }
      atomicAdd(&rowsq[srow], ss);
      uint4* dst = (uint4*)(&Gs[srow * SP + scol]);
      dst[0] = u.q[0]; dst[1] = u.q[1];
    }
    __syncthreads();

    // wave w computes rows 16w..16w+15 against all 8 column tiles
    int arow = (w << 4) + (lane & 15);
    int ak0 = (lane < 16) ? 0 : 8;  // A: interleaved K layout {k0..k0+7, k0+16..k0+23}
    union { v16bf v; uint4 q[2]; } ua;
    ua.q[0] = *(const uint4*)(&Qs[arow * SP + ak0]);
    ua.q[1] = *(const uint4*)(&Qs[arow * SP + ak0 + 16]);
    int bk0 = (lane < 16) ? 0 : 8;  // B: interleaved K layout {k0..k0+7, k0+16..k0+23}
#pragma unroll
    for (int tt = 0; tt < 8; tt++) {
      int brow = (tt << 4) + (lane & 15);
      union { v16bf v; uint4 q[2]; } ub;
      const uint4* bp = (const uint4*)(&Gs[brow * SP + bk0]);
      ub.q[0] = bp[0]; ub.q[1] = bp[2];
      acc[tt] = __builtin_amdgcn_wmma_f32_16x16x32_bf16(
          false, ua.v, false, ub.v, (short)0, acc[tt], false, false);
    }
    __syncthreads();
  }

  // gallery row inverse norms (scale applied at reduction; max commutes with >0 scale)
  if (t < 128) rowsq[t] = 1.f / fmaxf(sqrtf(rowsq[t]), EPSN);
  __syncthreads();

  int colS = lane & 15;
  int hi = lane >> 4;
  float rmax[8];
#pragma unroll
  for (int j = 0; j < 8; j++) rmax[j] = -3.4e38f;
#pragma unroll
  for (int tt = 0; tt < 8; tt++) {
    float s = rowsq[tt * 16 + colS];
    float cmax = -3.4e38f;
#pragma unroll
    for (int j = 0; j < 8; j++) {
      float v = acc[tt][j] * s;  // row = 16w + j + 8*hi, col = 16*tt + colS
      rmax[j] = fmaxf(rmax[j], v);
      cmax = fmaxf(cmax, v);
    }
    cmax = fmaxf(cmax, __shfl_xor(cmax, 16, 32));  // max over this wave's 16 rows
    if (hi == 0) colpart[w][tt * 16 + colS] = cmax;
  }
#pragma unroll
  for (int j = 0; j < 8; j++) {
#pragma unroll
    for (int m = 1; m < 16; m <<= 1)
      rmax[j] = fmaxf(rmax[j], __shfl_xor(rmax[j], m, 32));  // max over 16 cols in half
  }
  float s8 = 0.f;
#pragma unroll
  for (int j = 0; j < 8; j++) s8 += rmax[j];
  s8 += __shfl_xor(s8, 16, 32);  // both 8-row halves -> wave's 16 row-maxes summed
  if (lane == 0) atomicAdd(&sumRow, s8);
  __syncthreads();

  if (t < 128) {
    float m = colpart[0][t];
#pragma unroll
    for (int w2 = 1; w2 < 8; w2++) m = fmaxf(m, colpart[w2][t]);
    cm[t] = m;
  }
  __syncthreads();
  for (int st = 64; st > 0; st >>= 1) { if (t < st) cm[t] += cm[t + st]; __syncthreads(); }
  if (t == 0) {
    float emd = 0.5f * (sumRow + cm[0]) * (1.f / 128.f);
    float dc = dall[b * Gg + g];
    dist[b * CLSK + kk] = 0.6f * dc + 0.4f * emd;
  }
}

// ---------------- top-10 re-rank ----------------
__global__ void topk_final_kernel(const float* __restrict__ dist, const int* __restrict__ clsIdx,
                                  int* __restrict__ nb) {
  __shared__ float sd[CLSK];
  __shared__ float rv[128];
  __shared__ int   ri[128];
  int b = blockIdx.x, t = threadIdx.x;  // 128 threads
  if (t < CLSK) sd[t] = dist[b * CLSK + t];
  __syncthreads();
  for (int j = 0; j < TOPK; j++) {
    float bv = (t < CLSK) ? sd[t] : -3.4e38f;
    int bi = (t < CLSK) ? t : 0x7fffffff;
    rv[t] = bv; ri[t] = bi; __syncthreads();
    for (int st = 64; st > 0; st >>= 1) {
      if (t < st) {
        float ov = rv[t + st]; int oi = ri[t + st];
        if (ov > rv[t] || (ov == rv[t] && oi < ri[t])) { rv[t] = ov; ri[t] = oi; }
      }
      __syncthreads();
    }
    if (t == 0) { nb[b * TOPK + j] = clsIdx[b * CLSK + ri[0]]; sd[ri[0]] = -3.4e38f; }
    __syncthreads();
  }
}

// ---------------- gather raw gallery patches ----------------
__global__ void gather_kernel(const float* __restrict__ gpatch, const int* __restrict__ nb,
                              float* __restrict__ out) {
  int bj = blockIdx.x;  // 0..159
  int b = bj / TOPK, j = bj % TOPK;
  int chunk = blockIdx.y;  // 0..23
  int g = nb[b * TOPK + j];
  const float4* src = (const float4*)(gpatch + (size_t)g * Np * Cc);
  float4* dst = (float4*)(out + ((size_t)b * TOPK + j) * Np * Cc);
  int base = chunk * 1024 + threadIdx.x;
#pragma unroll
  for (int i = 0; i < 4; i++) {
    int idx = base + i * 256;
    dst[idx] = src[idx];
  }
}

// ---------------- mean of raw CLS vectors ----------------
__global__ void cls_mean_kernel(const float* __restrict__ gcls, const int* __restrict__ nb,
                                float* __restrict__ out2) {
  int b = blockIdx.x, t = threadIdx.x;
  for (int c = t; c < Cc; c += 256) {
    float s = 0.f;
    for (int j = 0; j < TOPK; j++) s += gcls[(size_t)nb[b * TOPK + j] * Cc + c];
    out2[(size_t)b * Cc + c] = s * (1.f / TOPK);
  }
}

extern "C" void kernel_launch(void* const* d_in, const int* in_sizes, int n_in,
                              void* d_out, int out_size, void* d_ws, size_t ws_size,
                              hipStream_t stream) {
  (void)in_sizes; (void)n_in; (void)out_size; (void)ws_size;
  const float* qc     = (const float*)d_in[0];
  const float* qp     = (const float*)d_in[1];
  const float* gcls   = (const float*)d_in[2];
  const float* gpatch = (const float*)d_in[3];

  char* ws = (char*)d_ws;
  float* dall   = (float*)(ws + 0);        // 16*2000*4 = 128000
  float* qcinv  = (float*)(ws + 128000);   // 64
  int*   clsIdx = (int*)  (ws + 128064);   // 6400
  float* dist   = (float*)(ws + 134464);   // 6400
  int*   nb     = (int*)  (ws + 140864);   // 640
  __bf16* Qn    = (__bf16*)(ws + 141504);  // 16*128*768*2 = 3145728

  float* out  = (float*)d_out;
  float* out2 = out + (size_t)Bq * TOPK * Np * Cc;

  qcinv_kernel<<<Bq, 256, 0, stream>>>(qc, qcinv);
  qnorm_kernel<<<Bq * Np, 256, 0, stream>>>(qp, Qn);
  cls_dist_kernel<<<Gg, 256, 0, stream>>>(gcls, qc, qcinv, dall);
  topk_cls_kernel<<<Bq, 256, 0, stream>>>(dall, clsIdx);
  emd_kernel<<<dim3(CLSK, Bq), 256, 0, stream>>>(gpatch, Qn, dall, clsIdx, dist);
  topk_final_kernel<<<Bq, 128, 0, stream>>>(dist, clsIdx, nb);
  gather_kernel<<<dim3(Bq * TOPK, 24), 256, 0, stream>>>(gpatch, nb, out);
  cls_mean_kernel<<<Bq, 256, 0, stream>>>(gcls, nb, out2);
}
